// NeRFLoss_50087908606650
// MI455X (gfx1250) — matrix operand order repaired and checked by reading, not compiled, not run on hardware
//
#include <hip/hip_runtime.h>
#include <hip/hip_bf16.h>

// ---------------------------------------------------------------------------
// Problem constants (match reference)
// ---------------------------------------------------------------------------
constexpr int N_RAYS  = 65536;
constexpr int S       = 64;
constexpr int NS      = N_RAYS * S;        // 4194304 samples
constexpr int RGB_LEN = 32768;

// Output layout (concatenated, float32)
constexpr int OFF_RGB   = 0;                       // 98304
constexpr int OFF_DLOSS = RGB_LEN * 3;             // 98304  .. +32768
constexpr int OFF_DREG  = OFF_DLOSS + (N_RAYS - RGB_LEN);   // 131072 .. +NS
constexpr int OFF_OPA   = OFF_DREG + NS;           // 4325376 .. +32768
constexpr int OFF_DIST  = OFF_OPA + RGB_LEN;       // 4358144 .. +65536

typedef __attribute__((ext_vector_type(16))) _Float16 v16h;
typedef __attribute__((ext_vector_type(8)))  float    v8f;
typedef __attribute__((ext_vector_type(4)))  unsigned int v4u;
typedef __attribute__((ext_vector_type(8)))  int      v8i;
typedef __attribute__((ext_vector_type(4)))  int      v4i;

#if defined(__has_builtin)
#  if __has_builtin(__builtin_amdgcn_wmma_f32_16x16x32_f16)
#    define HAVE_WMMA 1
#  endif
#  if __has_builtin(__builtin_amdgcn_tensor_load_to_lds) && \
      __has_builtin(__builtin_amdgcn_s_wait_tensorcnt)
#    define HAVE_TDM 1
#  endif
#endif

// ---------------------------------------------------------------------------
// Kernel 1: elementwise outputs (rgb_loss, depth_reg, opacity_loss)
// Fully coalesced float4 pass over the big arrays.
// ---------------------------------------------------------------------------
__global__ __launch_bounds__(256)
void nerf_eltwise(const float* __restrict__ rgb,
                  const float* __restrict__ trgb,
                  const float* __restrict__ ws,
                  const unsigned char* __restrict__ mask,
                  const float* __restrict__ opacity,
                  float* __restrict__ out) {
  const int i = blockIdx.x * 256 + threadIdx.x;   // float4 index, grid covers NS/4

  // depth_reg = 0.01 * where(mask, ws*ws, 0)
  {
    const float4 w = ((const float4*)ws)[i];
    const uchar4 m = ((const uchar4*)mask)[i];
    float4 r;
    r.x = m.x ? 0.01f * w.x * w.x : 0.0f;
    r.y = m.y ? 0.01f * w.y * w.y : 0.0f;
    r.z = m.z ? 0.01f * w.z * w.z : 0.0f;
    r.w = m.w ? 0.01f * w.w * w.w : 0.0f;
    ((float4*)(out + OFF_DREG))[i] = r;
  }
  // rgb_loss = (rgb[:RGB_LEN] - target_rgb)^2   (98304 elems = 24576 float4)
  if (i < (RGB_LEN * 3) / 4) {
    const float4 a = ((const float4*)rgb)[i];
    const float4 b = ((const float4*)trgb)[i];
    float4 r;
    r.x = (a.x - b.x) * (a.x - b.x);
    r.y = (a.y - b.y) * (a.y - b.y);
    r.z = (a.z - b.z) * (a.z - b.z);
    r.w = (a.w - b.w) * (a.w - b.w);
    ((float4*)(out + OFF_RGB))[i] = r;
  }
  // opacity_loss = 0.001 * (-o * log(o)), o = opacity[:RGB_LEN] + 1e-10
  if (i < RGB_LEN / 4) {
    const float4 o = ((const float4*)opacity)[i];
    float4 r;
    float ox = o.x + 1e-10f; r.x = 0.001f * (-ox * logf(ox));
    float oy = o.y + 1e-10f; r.y = 0.001f * (-oy * logf(oy));
    float oz = o.z + 1e-10f; r.z = 0.001f * (-oz * logf(oz));
    float ow = o.w + 1e-10f; r.w = 0.001f * (-ow * logf(ow));
    ((float4*)(out + OFF_OPA))[i] = r;
  }
}

// ---------------------------------------------------------------------------
// TDM helpers: wave-level DMA of a 16-ray x 64-sample f32 tile into LDS,
// with pad_enable giving a 68-float LDS row pitch (272B) for bank spreading.
// ---------------------------------------------------------------------------
#if defined(HAVE_TDM)
__device__ __forceinline__ unsigned lds_byte_offset(const void* p) {
  // generic -> LDS address space; as(3) pointer value IS the byte offset.
  return (unsigned)(unsigned long long)(__attribute__((address_space(3))) const char*)p;
}

__device__ __forceinline__ void tdm_load_tile16x64(const float* gptr, unsigned lds_off) {
  const unsigned long long ga = (unsigned long long)gptr;
  v4u g0;
  g0[0] = 1u;                                   // count=1, user descriptor
  g0[1] = lds_off;                              // lds_addr (bytes)
  g0[2] = (unsigned)(ga & 0xFFFFFFFFull);       // global_addr[31:0]
  g0[3] = (unsigned)((ga >> 32) & 0x1FFFFFFull) // global_addr[56:32]
        | (2u << 30);                           // type=2 ("image")
  v8i g1;
  g1[0] = (2 << 16)        // data_size = 4B
        | (1 << 20)        // pad_enable
        | (5 << 22)        // pad_interval: 64 dwords (256B = one row)
        | (3 << 25);       // pad_amount: 4 dwords (16B) -> pitch 272B
  g1[1] = (64 << 16);      // tensor_dim0 = 64 (bits[79:48], low 16)
  g1[2] = 0;               // dim0 hi | tensor_dim1 low16 (65536 & 0xffff = 0)
  g1[3] = 1 | (64 << 16);  // tensor_dim1 hi16 = 1 (=65536), tile_dim0 = 64
  g1[4] = 16;              // tile_dim1 = 16 rays, tile_dim2 = 0
  g1[5] = 64;              // tensor_dim0_stride = 64 (low 32)
  g1[6] = 0;
  g1[7] = 0;
  const v4i z4 = {0, 0, 0, 0};
  const v8i z8 = {0, 0, 0, 0, 0, 0, 0, 0};
  // 6-arg form (this toolchain): (g0, g1, g2, g3, g4, cpol)
  __builtin_amdgcn_tensor_load_to_lds(g0, g1, z4, z4, z8, 0);
}
#endif

// ---------------------------------------------------------------------------
// Kernel 2: per-ray exclusive scans + distortion + depth loss.
// One wave handles 16 rays (B-matrix columns). Exclusive scans of ws and
// ws*ts are done on the matrix pipe: D = A_tri x B_data + C_carry with
// hi/lo f16 splitting for exact-to-f32 accumulation.
// ---------------------------------------------------------------------------
#define WAVES_PER_BLOCK 4
#define TILE_PITCH 68   // floats per ray row in LDS (64 + 4 pad)

__global__ __launch_bounds__(128)
void nerf_scan(const float* __restrict__ ws,
               const float* __restrict__ ts,
               const float* __restrict__ deltas,
               const float* __restrict__ target_depth,
               float* __restrict__ out) {
  __shared__ float tile[WAVES_PER_BLOCK][3][16 * TILE_PITCH];

  const int tid  = threadIdx.x;
  const int wave = tid >> 5;
  const int lane = tid & 31;
  const int n    = lane & 15;   // ray-in-tile (B/C/D column), also A row M
  const int half = lane >> 4;   // 0: samples 0..15 of block, 1: samples 16..31
  const int rayBase = (blockIdx.x * WAVES_PER_BLOCK + wave) * 16;

  const float* gsrc[3] = {ws, ts, deltas};

#if defined(HAVE_TDM)
  #pragma unroll
  for (int a = 0; a < 3; ++a)
    tdm_load_tile16x64(gsrc[a] + (size_t)rayBase * S, lds_byte_offset(&tile[wave][a][0]));
  __builtin_amdgcn_s_wait_tensorcnt(0);
#else
  #pragma unroll
  for (int a = 0; a < 3; ++a) {
    const float4* g4 = (const float4*)(gsrc[a] + (size_t)rayBase * S);
    for (int q = lane; q < 256; q += 32) {      // 256 float4 = 16 rays x 64 f32
      const int r = q >> 4, c = q & 15;
      *(float4*)&tile[wave][a][r * TILE_PITCH + c * 4] = g4[r * 16 + c];
    }
  }
#endif
  __syncthreads();

  // A matrix: A[m,k] = 1 if k < 2m (strict lower-triangular at even stride).
  // 16-bit A 16x32 layout (contiguous, matching B): lane<16: k=e;
  //                        lane>=16: k=e+16.
  v16h atri;
  #pragma unroll
  for (int e = 0; e < 16; ++e) {
    const int k = e + (half ? 16 : 0);
    atri[e] = (k < 2 * n) ? (_Float16)1.0f : (_Float16)0.0f;
  }

  float carryW = 0.0f, carryWT = 0.0f;   // identical in lanes n and n+16
  float distAcc = 0.0f, depthAcc = 0.0f;

  #pragma unroll
  for (int b = 0; b < 2; ++b) {
    const int base = n * TILE_PITCH + b * 32 + half * 16;
    float wsv[16], tsv[16], dlv[16], wtv[16];
    #pragma unroll
    for (int q = 0; q < 4; ++q) {
      const float4 a = *(const float4*)&tile[wave][0][base + q * 4];
      const float4 t = *(const float4*)&tile[wave][1][base + q * 4];
      const float4 d = *(const float4*)&tile[wave][2][base + q * 4];
      wsv[q*4+0]=a.x; wsv[q*4+1]=a.y; wsv[q*4+2]=a.z; wsv[q*4+3]=a.w;
      tsv[q*4+0]=t.x; tsv[q*4+1]=t.y; tsv[q*4+2]=t.z; tsv[q*4+3]=t.w;
      dlv[q*4+0]=d.x; dlv[q*4+1]=d.y; dlv[q*4+2]=d.z; dlv[q*4+3]=d.w;
    }

    // hi/lo f16 split of ws and wt = ws*ts (exact products vs the ones matrix)
    v16h bwh, bwl, bth, btl;
    float swLocal = 0.0f, swtLocal = 0.0f;
    #pragma unroll
    for (int j = 0; j < 16; ++j) {
      const float w = wsv[j];
      const _Float16 h = (_Float16)w;
      bwh[j] = h; bwl[j] = (_Float16)(w - (float)h);
      const float wt = w * tsv[j];
      wtv[j] = wt;
      const _Float16 h2 = (_Float16)wt;
      bth[j] = h2; btl[j] = (_Float16)(wt - (float)h2);
      swLocal += w; swtLocal += wt;
    }
    const float swOther  = __shfl_xor(swLocal, 16, 32);
    const float swtOther = __shfl_xor(swtLocal, 16, 32);

    float exw_e[8], exwt_e[8];   // exclusive prefix at even positions 2j
#if defined(HAVE_WMMA)
    v8f cw, cwt;
    #pragma unroll
    for (int j = 0; j < 8; ++j) { cw[j] = carryW; cwt[j] = carryWT; }
    cw  = __builtin_amdgcn_wmma_f32_16x16x32_f16(false, atri, false, bwh, (short)0, cw,  false, false);
    cw  = __builtin_amdgcn_wmma_f32_16x16x32_f16(false, atri, false, bwl, (short)0, cw,  false, false);
    cwt = __builtin_amdgcn_wmma_f32_16x16x32_f16(false, atri, false, bth, (short)0, cwt, false, false);
    cwt = __builtin_amdgcn_wmma_f32_16x16x32_f16(false, atri, false, btl, (short)0, cwt, false, false);
    #pragma unroll
    for (int j = 0; j < 8; ++j) { exw_e[j] = cw[j]; exwt_e[j] = cwt[j]; }
#else
    // fallback: lane-local exact exclusive scan
    float runW = carryW  + (half ? swOther  : 0.0f);
    float runT = carryWT + (half ? swtOther : 0.0f);
    #pragma unroll
    for (int j = 0; j < 16; ++j) {
      if ((j & 1) == 0) { exw_e[j >> 1] = runW; exwt_e[j >> 1] = runT; }
      runW += wsv[j]; runT += wtv[j];
    }
    (void)atri;
#endif

    // per-sample distortion terms (lane-local: D hands us exactly our window)
    #pragma unroll
    for (int j = 0; j < 8; ++j) {
      const float exw = exw_e[j], exwt = exwt_e[j];
      const float w0 = wsv[2*j], t0 = tsv[2*j], d0 = dlv[2*j];
      distAcc += 2.0f * w0 * (t0 * exw - exwt) + w0 * w0 * d0 * (1.0f / 3.0f);
      const float exw1  = exw  + w0;
      const float exwt1 = exwt + wtv[2*j];
      const float w1 = wsv[2*j+1], t1 = tsv[2*j+1], d1 = dlv[2*j+1];
      distAcc += 2.0f * w1 * (t1 * exw1 - exwt1) + w1 * w1 * d1 * (1.0f / 3.0f);
      depthAcc += wtv[2*j] + wtv[2*j+1];
    }

    carryW  += swLocal  + swOther;
    carryWT += swtLocal + swtOther;
  }

  // close per-ray reductions across the two half-lanes
  distAcc  += __shfl_xor(distAcc, 16, 32);
  depthAcc += __shfl_xor(depthAcc, 16, 32);

  if (half == 0) {
    const int ray = rayBase + n;
    out[OFF_DIST + ray] = 0.001f * distAcc;
    if (ray >= RGB_LEN) {
      const float diff = depthAcc - target_depth[ray - RGB_LEN];
      out[OFF_DLOSS + (ray - RGB_LEN)] = 0.01f * diff * diff;
    }
  }
}

// ---------------------------------------------------------------------------
// Launch
// ---------------------------------------------------------------------------
extern "C" void kernel_launch(void* const* d_in, const int* in_sizes, int n_in,
                              void* d_out, int out_size, void* d_ws, size_t ws_size,
                              hipStream_t stream) {
  const float* rgb      = (const float*)d_in[0];
  const float* trgb     = (const float*)d_in[1];
  const float* ws       = (const float*)d_in[2];
  const float* ts       = (const float*)d_in[3];
  const float* deltas   = (const float*)d_in[4];
  const float* opacity  = (const float*)d_in[5];
  const unsigned char* mask = (const unsigned char*)d_in[6];
  const float* tdepth   = (const float*)d_in[8];
  float* out = (float*)d_out;

  // Elementwise pass: NS/4 float4s / 256 threads
  nerf_eltwise<<<(NS / 4) / 256, 256, 0, stream>>>(rgb, trgb, ws, mask, opacity, out);
  // Scan pass: 16 rays/wave, 4 waves/block -> 64 rays/block
  nerf_scan<<<N_RAYS / (16 * WAVES_PER_BLOCK), 128, 0, stream>>>(ws, ts, deltas, tdepth, out);
}
